// RVAEEncodeToLatentModel_69252052681252
// MI455X (gfx1250) — matrix operand order repaired and checked by reading, not compiled, not run on hardware
//
#include <hip/hip_runtime.h>
#include <hip/hip_bf16.h>

// ---------------------------------------------------------------------------
// CDNA5 (gfx1250) wave32 WMMA implementation of:
//   conv3x3(SAME) -> permute -> linear -> VQ nearest-codebook argmin
// Matrix stages: v_wmma_f32_16x16x32_bf16.
// B-tiles shared by all 8 waves of a block are staged in LDS with
// global_load_async_to_lds_b128 (ASYNCcnt) and double buffering.
// ---------------------------------------------------------------------------

typedef __attribute__((ext_vector_type(16))) __bf16 v16bf;
typedef __attribute__((ext_vector_type(8)))  float  v8f;

union Frag {            // 16 bf16 lane-operand for WMMA == two 16B chunks
    v16bf v;
    uint4 q[2];
};

__device__ __forceinline__ unsigned short f32_to_bf16_rne(float f) {
    unsigned int u = __float_as_uint(f);
    unsigned int lsb = (u >> 16) & 1u;
    u += 0x7fffu + lsb;                 // round to nearest even
    return (unsigned short)(u >> 16);
}

// Async global->LDS 128-bit copy (CDNA5, tracked by ASYNCcnt).
// ldsoff: wave-relative LDS byte offset (low 32 bits of flat shared ptr).
#define ASYNC_LOAD_B128(ldsoff_u32, gaddr_u64)                         \
    asm volatile("global_load_async_to_lds_b128 %0, %1, off"           \
                 :: "v"(ldsoff_u32), "v"(gaddr_u64) : "memory")

#define WAIT_ASYNCCNT0() asm volatile("s_wait_asynccnt 0x0" ::: "memory")

__device__ __forceinline__ unsigned lds_off(const void* p) {
    return (unsigned)(unsigned long long)p;   // low 32 bits = LDS offset
}

#define B_   256
#define CIN_ 512
#define COUT_ 256
#define HW_  8
#define K_   8192
#define MTOT (B_ * HW_ * HW_)          // 16384 pixel rows

// --------------------------- prep kernels ----------------------------------

// NCHW f32 -> NHWC bf16 (A rows of conv GEMM become contiguous in CIN)
__global__ void prep_latent(const float* __restrict__ in, unsigned short* __restrict__ out) {
    int i = blockIdx.x * blockDim.x + threadIdx.x;
    if (i >= B_ * HW_ * HW_ * CIN_) return;
    int c = i & (CIN_ - 1);
    int r = i >> 9;
    int w = r & 7; r >>= 3;
    int h = r & 7;
    int b = r >> 3;
    out[i] = f32_to_bf16_rne(in[((b * CIN_ + c) * HW_ + h) * HW_ + w]);
}

// conv_w [COUT][CIN][3][3] f32 -> Wp [9][COUT][CIN] bf16 (B rows contiguous in K)
__global__ void prep_convw(const float* __restrict__ in, unsigned short* __restrict__ out) {
    int i = blockIdx.x * blockDim.x + threadIdx.x;
    if (i >= 9 * COUT_ * CIN_) return;
    int cin = i & (CIN_ - 1);
    int co  = (i >> 9) & (COUT_ - 1);
    int tap = i >> 17;
    out[i] = f32_to_bf16_rne(in[(co * CIN_ + cin) * 9 + tap]);
}

// lin_w f32 -> bf16 (already [D][C], K-contiguous)
__global__ void prep_linw(const float* __restrict__ in, unsigned short* __restrict__ out) {
    int i = blockIdx.x * blockDim.x + threadIdx.x;
    if (i >= COUT_ * COUT_) return;
    out[i] = f32_to_bf16_rne(in[i]);
}

// emb f32 -> bf16 copy + per-code squared norm (f32)
__global__ void prep_emb(const float* __restrict__ emb,
                         unsigned short* __restrict__ embb,
                         float* __restrict__ enorm) {
    __shared__ float red[256];
    int k = blockIdx.x;
    int t = threadIdx.x;
    float v = emb[k * COUT_ + t];
    embb[k * COUT_ + t] = f32_to_bf16_rne(v);
    red[t] = v * v;
    __syncthreads();
    for (int s = 128; s > 0; s >>= 1) {
        if (t < s) red[t] += red[t + s];
        __syncthreads();
    }
    if (t == 0) enorm[k] = red[0];
}

// ------------------------- conv implicit GEMM ------------------------------
// One wave == one strip of 16 pixels (2 image rows). M=16, N=256 (16 tiles),
// K=4608 (9 taps x 16 k-steps of 32). The per-step B slab (all 256 couts x
// 32 k = 16 KB) is identical for every wave in the block -> stage it in LDS
// via async copies, double-buffered across the 144 k-steps.
#define CW_STRIDE 40   // halfs per LDS row (32 payload + 8 pad -> conflict-free)

__global__ void __launch_bounds__(256)
conv_gemm(const unsigned short* __restrict__ X,   // NHWC bf16
          const unsigned short* __restrict__ Wp,  // [9][COUT][CIN] bf16
          const float* __restrict__ bias,
          unsigned short* __restrict__ Y) {       // [MTOT][COUT] bf16
    alignas(16) __shared__ unsigned short wTile[2][COUT_ * CW_STRIDE]; // 2 x 20 KB

    const int tid   = threadIdx.x;
    const int lane  = tid & 31;
    const int wave  = tid >> 5;
    const int strip = blockIdx.x * 8 + wave;
    const int mbase = strip * 16;
    const int b  = mbase >> 6;          // 64 pixels per image
    const int h0 = (mbase & 63) >> 3;   // first of two image rows
    const int rowm = lane & 15;         // A-matrix row for this lane
    const bool hiK = lane >= 16;        // upper K half per ISA A/B layouts
    const int h = h0 + (rowm >> 3);
    const int w = rowm & 7;
    const int ncol = lane & 15;

    v8f acc[16];
    const v8f vz = {0.f, 0.f, 0.f, 0.f, 0.f, 0.f, 0.f, 0.f};
#pragma unroll
    for (int nt = 0; nt < 16; ++nt) acc[nt] = vz;

    // stage s = tap*16 + ks ; each thread copies one cout row (4 x 16B)
    auto issue_stage = [&](int s, int buf) {
        const int tap = s >> 4, ks = s & 15;
        const unsigned short* src = Wp + tap * COUT_ * CIN_ + tid * CIN_ + ks * 32;
#pragma unroll
        for (int i = 0; i < 4; ++i) {
            unsigned lo = lds_off(&wTile[buf][tid * CW_STRIDE + i * 8]);
            unsigned long long ga = (unsigned long long)(src + i * 8);
            ASYNC_LOAD_B128(lo, ga);
        }
    };

    issue_stage(0, 0);
    WAIT_ASYNCCNT0();
    __syncthreads();

    for (int s = 0; s < 144; ++s) {
        if (s + 1 < 144) issue_stage(s + 1, (s + 1) & 1);

        const int tap = s >> 4, ks = s & 15;
        const int dh = tap / 3 - 1, dw = tap % 3 - 1;
        const int hh = h + dh, ww = w + dw;
        const bool valid = (hh >= 0) && (hh < 8) && (ww >= 0) && (ww < 8);

        Frag aF;
        if (valid) {
            const unsigned short* arow =
                X + (((b * HW_ + hh) * HW_ + ww) * CIN_) + ks * 32 + (hiK ? 8 : 0);
            aF.q[0] = *(const uint4*)(arow);
            aF.q[1] = *(const uint4*)(arow + 16);
        } else {
            aF.q[0] = make_uint4(0u, 0u, 0u, 0u);
            aF.q[1] = make_uint4(0u, 0u, 0u, 0u);
        }

        const unsigned short* wbuf = &wTile[s & 1][0];
#pragma unroll
        for (int nt = 0; nt < 16; ++nt) {
            const unsigned short* brow =
                wbuf + (nt * 16 + ncol) * CW_STRIDE + (hiK ? 8 : 0);
            Frag bF;
            bF.q[0] = *(const uint4*)(brow);
            bF.q[1] = *(const uint4*)(brow + 16);
            acc[nt] = __builtin_amdgcn_wmma_f32_16x16x32_bf16(
                false, aF.v, false, bF.v, (short)0, acc[nt], false, false);
        }

        WAIT_ASYNCCNT0();
        __syncthreads();
    }

    // epilogue: +bias, bf16 store. C layout: VGPR r -> M=r (lanes 0-15) / M=8+r.
    const int mloc = hiK ? 8 : 0;
#pragma unroll
    for (int nt = 0; nt < 16; ++nt) {
        const float bv = bias[nt * 16 + ncol];
#pragma unroll
        for (int r = 0; r < 8; ++r) {
            Y[(mbase + mloc + r) * COUT_ + nt * 16 + ncol] =
                f32_to_bf16_rne(acc[nt][r] + bv);
        }
    }
}

// ----------------------------- linear GEMM ---------------------------------
__global__ void __launch_bounds__(256)
lin_gemm(const unsigned short* __restrict__ Yin,  // [MTOT][COUT] bf16
         const unsigned short* __restrict__ Wl,   // [COUT][COUT] bf16
         const float* __restrict__ bias,
         unsigned short* __restrict__ Q) {        // [MTOT][COUT] bf16
    const int lane  = threadIdx.x & 31;
    const int wave  = threadIdx.x >> 5;
    const int strip = blockIdx.x * 8 + wave;
    const int mbase = strip * 16;
    const int rowm  = lane & 15;
    const bool hiK  = lane >= 16;
    const int ncol  = lane & 15;

    v8f acc[16];
    const v8f vz = {0.f, 0.f, 0.f, 0.f, 0.f, 0.f, 0.f, 0.f};
#pragma unroll
    for (int nt = 0; nt < 16; ++nt) acc[nt] = vz;

    const unsigned short* arow = Yin + (mbase + rowm) * COUT_;
    for (int ks = 0; ks < COUT_ / 32; ++ks) {
        const int k0 = ks * 32;
        Frag aF;
        const int off = k0 + (hiK ? 8 : 0);
        aF.q[0] = *(const uint4*)(arow + off);
        aF.q[1] = *(const uint4*)(arow + off + 16);
#pragma unroll
        for (int nt = 0; nt < 16; ++nt) {
            const unsigned short* brow =
                Wl + (nt * 16 + ncol) * COUT_ + k0 + (hiK ? 8 : 0);
            Frag bF;
            bF.q[0] = *(const uint4*)(brow);
            bF.q[1] = *(const uint4*)(brow + 16);
            acc[nt] = __builtin_amdgcn_wmma_f32_16x16x32_bf16(
                false, aF.v, false, bF.v, (short)0, acc[nt], false, false);
        }
    }

    const int mloc = hiK ? 8 : 0;
#pragma unroll
    for (int nt = 0; nt < 16; ++nt) {
        const float bv = bias[nt * 16 + ncol];
#pragma unroll
        for (int r = 0; r < 8; ++r) {
            Q[(mbase + mloc + r) * COUT_ + nt * 16 + ncol] =
                f32_to_bf16_rne(acc[nt][r] + bv);
        }
    }
}

// ------------------------- VQ distance + argmin ----------------------------
// argmin_k( |e_k|^2 - 2 x.e_k )  -- ||x||^2 dropped (constant per row).
// A fragments (16 pixels x 256 feats) stay in registers across all 8192
// codes; the 16-code x 256-feat codebook tile (8 KB) is shared by all 8
// waves -> async-staged in LDS, double-buffered across the 512 tiles.
#define VQ_STRIDE 264  // halfs per LDS row (256 payload + 8 pad -> conflict-free)

__global__ void __launch_bounds__(256)
vq_argmin(const unsigned short* __restrict__ Q,   // [MTOT][COUT] bf16
          const unsigned short* __restrict__ E,   // [K][COUT] bf16
          const float* __restrict__ En,           // [K] f32
          int* __restrict__ out) {                // [MTOT]
    alignas(16) __shared__ unsigned short eTile[2][16 * VQ_STRIDE]; // 2 x 8.25 KB

    const int tid   = threadIdx.x;
    const int lane  = tid & 31;
    const int wave  = tid >> 5;
    const int strip = blockIdx.x * 8 + wave;
    const int mbase = strip * 16;
    const int rowm  = lane & 15;
    const bool hiK  = lane >= 16;
    const int ncol  = lane & 15;

    Frag aF[8];
    const unsigned short* arow = Q + (mbase + rowm) * COUT_;
#pragma unroll
    for (int ks = 0; ks < 8; ++ks) {
        const int off = ks * 32 + (hiK ? 8 : 0);
        aF[ks].q[0] = *(const uint4*)(arow + off);
        aF[ks].q[1] = *(const uint4*)(arow + off + 16);
    }

    // tile ct = 16 codes x 256 halfs = 512 x 16B chunks; 2 chunks per thread
    auto issue_tile = [&](int ct, int buf) {
#pragma unroll
        for (int j = 0; j < 2; ++j) {
            const int c   = tid + j * 256;
            const int row = c >> 5;        // 0..15
            const int col = c & 31;        // x8 halfs
            unsigned lo = lds_off(&eTile[buf][row * VQ_STRIDE + col * 8]);
            unsigned long long ga =
                (unsigned long long)(E + (ct * 16 + row) * COUT_ + col * 8);
            ASYNC_LOAD_B128(lo, ga);
        }
    };

    float best[8];
    int   bidx[8];
#pragma unroll
    for (int r = 0; r < 8; ++r) { best[r] = 3.4e38f; bidx[r] = 0; }

    issue_tile(0, 0);
    WAIT_ASYNCCNT0();
    __syncthreads();

    const v8f vz = {0.f, 0.f, 0.f, 0.f, 0.f, 0.f, 0.f, 0.f};
    for (int ct = 0; ct < K_ / 16; ++ct) {
        if (ct + 1 < K_ / 16) issue_tile(ct + 1, (ct + 1) & 1);

        const unsigned short* erow =
            &eTile[ct & 1][ncol * VQ_STRIDE + (hiK ? 8 : 0)];
        auto load_b = [&](int ks, Frag& f) {
            f.q[0] = *(const uint4*)(erow + ks * 32);
            f.q[1] = *(const uint4*)(erow + ks * 32 + 16);
        };

        Frag bF[2];
        load_b(0, bF[0]);
        v8f acc = vz;
#pragma unroll
        for (int ks = 0; ks < 8; ++ks) {
            if (ks + 1 < 8) load_b(ks + 1, bF[(ks + 1) & 1]);
            acc = __builtin_amdgcn_wmma_f32_16x16x32_bf16(
                false, aF[ks].v, false, bF[ks & 1].v, (short)0, acc, false, false);
        }
        const int n = ct * 16 + ncol;
        const float en = En[n];
#pragma unroll
        for (int r = 0; r < 8; ++r) {
            const float d = en - 2.0f * acc[r];
            if (d < best[r]) { best[r] = d; bidx[r] = n; }
        }

        WAIT_ASYNCCNT0();
        __syncthreads();
    }

    // Cross-lane argmin within each 16-lane group (lanes 0-15: M=r, 16-31: M=8+r).
#pragma unroll
    for (int mask = 8; mask > 0; mask >>= 1) {
#pragma unroll
        for (int r = 0; r < 8; ++r) {
            const float ov = __shfl_xor(best[r], mask, 32);
            const int   oi = __shfl_xor(bidx[r], mask, 32);
            if (ov < best[r] || (ov == best[r] && oi < bidx[r])) {
                best[r] = ov; bidx[r] = oi;
            }
        }
    }
    if ((lane & 15) == 0) {
        const int mloc = hiK ? 8 : 0;
#pragma unroll
        for (int r = 0; r < 8; ++r) out[mbase + mloc + r] = bidx[r];
    }
}

// ------------------------------ launcher -----------------------------------
extern "C" void kernel_launch(void* const* d_in, const int* in_sizes, int n_in,
                              void* d_out, int out_size, void* d_ws, size_t ws_size,
                              hipStream_t stream) {
    const float* latent = (const float*)d_in[0];
    const float* conv_w = (const float*)d_in[1];
    const float* conv_b = (const float*)d_in[2];
    const float* lin_w  = (const float*)d_in[3];
    const float* lin_b  = (const float*)d_in[4];
    const float* emb    = (const float*)d_in[5];
    int* out = (int*)d_out;

    // workspace carve-up (bf16 stored as u16). qbuf aliases latnhwc: last
    // reader of latnhwc (conv_gemm) completes before lin_gemm writes qbuf
    // (stream-ordered), so the alias is safe.
    char* ws = (char*)d_ws;
    unsigned short* latnhwc = (unsigned short*)ws;                       // 16.78 MB
    unsigned short* qbuf    = latnhwc;                                   // alias (8.39 MB)
    ws += (size_t)B_ * HW_ * HW_ * CIN_ * 2;
    unsigned short* wp   = (unsigned short*)ws;  ws += (size_t)9 * COUT_ * CIN_ * 2;  // 2.36 MB
    unsigned short* wl   = (unsigned short*)ws;  ws += (size_t)COUT_ * COUT_ * 2;     // 0.13 MB
    unsigned short* ybuf = (unsigned short*)ws;  ws += (size_t)MTOT * COUT_ * 2;      // 8.39 MB
    unsigned short* embb = (unsigned short*)ws;  ws += (size_t)K_ * COUT_ * 2;        // 4.19 MB
    float* enorm         = (float*)ws;           ws += (size_t)K_ * 4;                // 32 KB

    prep_latent<<<(B_ * HW_ * HW_ * CIN_) / 256, 256, 0, stream>>>(latent, latnhwc);
    prep_convw <<<(9 * COUT_ * CIN_) / 256, 256, 0, stream>>>(conv_w, wp);
    prep_linw  <<<(COUT_ * COUT_) / 256, 256, 0, stream>>>(lin_w, wl);
    prep_emb   <<<K_, 256, 0, stream>>>(emb, embb, enorm);

    // 1024 strips of 16 pixels; 8 waves (strips) per 256-thread block.
    conv_gemm<<<MTOT / 16 / 8, 256, 0, stream>>>(latnhwc, wp, conv_b, ybuf);
    lin_gemm <<<MTOT / 16 / 8, 256, 0, stream>>>(ybuf, wl, lin_b, qbuf);
    vq_argmin<<<MTOT / 16 / 8, 256, 0, stream>>>(qbuf, embb, enorm, out);
}
